// LinearChainCRF_73366631350650
// MI455X (gfx1250) — matrix operand order repaired and checked by reading, not compiled, not run on hardware
//
#include <hip/hip_runtime.h>
#include <cstddef>

// Problem constants (from reference): B=128, S=1024, T=256
#define B_DIM 128
#define S_DIM 1024
#define T_DIM 256

typedef __attribute__((ext_vector_type(16))) _Float16 v16h;
typedef __attribute__((ext_vector_type(8)))  float    v8f;

// ---------------- Dynamic LDS layout (bytes), all 32B aligned -------------
// Efrag: exp(transitions) pre-swizzled into WMMA B-fragment layout:
//        [kb(8)][nt(16)][lane(32)][half(16)]  f16  -> 131072 B
// Pfrag: exp(alpha - rowmax) in WMMA A-fragment layout:
//        [kb(8)][lane(32)][half(16)]          f16  ->   8192 B
// alpha: [16][ASTRIDE] f32 (padded rows)           ->  16896 B
// mrow : [16] f32                                  ->     64 B
#define EF_OFF   0
#define PF_OFF   131072
#define AL_OFF   139264
#define MR_OFF   156160
#define SMEM_BYTES 156224
#define ASTRIDE  264

// =====================================================================
// Kernel A: zero the output scalar and compute the joint (numerator)
// log-likelihood per batch element. Tiny: one thread per batch row.
// =====================================================================
__global__ void crf_num_init_kernel(const float* __restrict__ logits,
                                    const float* __restrict__ trans,
                                    const int*   __restrict__ tags,
                                    const int*   __restrict__ mask,
                                    float*       __restrict__ out,
                                    float*       __restrict__ ws_num) {
    int b = threadIdx.x;
    if (b == 0) out[0] = 0.0f;           // kernel A retires before kernel B starts
    if (b >= B_DIM) return;

    float sc = 0.0f;
    int   msum = 0;
    const int base = b * S_DIM;
    for (int s = 0; s < S_DIM - 1; ++s) {
        int t0 = tags[base + s];
        int t1 = tags[base + s + 1];
        int m0 = mask[base + s];
        int m1 = mask[base + s + 1];
        msum += m0;
        sc += trans[t0 * T_DIM + t1] * (float)m1;
        sc += logits[((size_t)base + s) * T_DIM + t0] * (float)m0;
    }
    int mL = mask[base + S_DIM - 1];
    msum += mL;
    int last_idx = msum - 1;
    if (last_idx < 0) last_idx = 0;
    sc += logits[((size_t)base + (S_DIM - 1)) * T_DIM + tags[base + last_idx]] * (float)mL;
    ws_num[b] = sc;
}

// =====================================================================
// Kernel B: forward algorithm (denominator) via f16 WMMA GEMM per step.
//   grid = 8 blocks (16 batch rows each), block = 256 threads = 8 waves.
//   Each wave owns 2 N-tiles (32 of 256 output tags).
// =====================================================================
__global__ __launch_bounds__(256) void crf_fwd_kernel(
        const float* __restrict__ logits,
        const float* __restrict__ trans,
        const int*   __restrict__ mask,
        float*       __restrict__ out,
        const float* __restrict__ ws_num) {
    extern __shared__ char smem[];
    _Float16* __restrict__ EfH   = reinterpret_cast<_Float16*>(smem + EF_OFF);
    _Float16* __restrict__ PfH   = reinterpret_cast<_Float16*>(smem + PF_OFF);
    float*    __restrict__ alphaS= reinterpret_cast<float*>(smem + AL_OFF);
    float*    __restrict__ mrowS = reinterpret_cast<float*>(smem + MR_OFF);
    const v16h* __restrict__ Ef16 = reinterpret_cast<const v16h*>(smem + EF_OFF);
    const v16h* __restrict__ Pf16 = reinterpret_cast<const v16h*>(smem + PF_OFF);

    const int tid  = threadIdx.x;
    const int lane = tid & 31;
    const int w    = tid >> 5;      // wave id 0..7
    const int m0   = blockIdx.x * 16;

    // ---- One-time: E = exp(transitions) pre-swizzled into B-fragment layout.
    // B (K=32 x N=16, f16): lane = n + 16*((k_local>>3)&1); half = (k_local&7) | ((k_local>>4)<<3).
    for (int idx = tid; idx < 8 * 16 * 32 * 16; idx += 256) {
        int h  = idx & 15;
        int ln = (idx >> 4) & 31;
        int nt = (idx >> 9) & 15;
        int kb = idx >> 13;
        int k  = kb * 32 + ((h >> 3) << 4) + ((ln >> 4) << 3) + (h & 7);
        int n  = nt * 16 + (ln & 15);
        EfH[idx] = (_Float16)__expf(trans[k * T_DIM + n]);
    }

    // ---- alpha[r][:] = logits[b, 0, :]
    {
        int r  = tid >> 4;
        int cb = tid & 15;
        const size_t rowbase = ((size_t)(m0 + r) * S_DIM) * T_DIM;
        for (int c = cb; c < T_DIM; c += 16)
            alphaS[r * ASTRIDE + c] = logits[rowbase + c];
        // prefetch step-1 emissions
        __builtin_prefetch(&logits[rowbase + T_DIM + cb * 16], 0, 1);
    }
    __syncthreads();

    const int nt0 = 2 * w, nt1 = 2 * w + 1;

    for (int s = 1; s < S_DIM; ++s) {
        // ---------- Phase 1: row max + P = exp(alpha - max), A-fragment pack
        {
            int r  = tid >> 4;
            int cb = tid & 15;
            float mx = -__builtin_inff();
            for (int c = cb; c < T_DIM; c += 16)
                mx = fmaxf(mx, alphaS[r * ASTRIDE + c]);
            #pragma unroll
            for (int off = 1; off < 16; off <<= 1)
                mx = fmaxf(mx, __shfl_xor(mx, off, 32));
            if (cb == 0) mrowS[r] = mx;
            for (int c = cb; c < T_DIM; c += 16) {
                float p  = __expf(alphaS[r * ASTRIDE + c] - mx);
                int kb   = c >> 5;
                int kl   = c & 31;
                int ln   = r + (((kl >> 3) & 1) << 4);       // A-layout lane
                int h    = (kl & 7) | (((kl >> 4) & 1) << 3); // A-layout half
                PfH[(kb * 32 + ln) * 16 + h] = (_Float16)p;
            }
            // prefetch next step's emission row segment
            if (s + 1 < S_DIM)
                __builtin_prefetch(
                    &logits[(((size_t)(m0 + r) * S_DIM) + (s + 1)) * T_DIM + cb * 16], 0, 1);
        }
        __syncthreads();

        // ---------- Phase 2: acc = P (16x256) @ E (256x32-slice) via WMMA
        v8f acc0 = {};
        v8f acc1 = {};
        #pragma unroll
        for (int kb = 0; kb < 8; ++kb) {
            v16h a  = Pf16[kb * 32 + lane];
            v16h b0 = Ef16[(kb * 16 + nt0) * 32 + lane];
            v16h b1 = Ef16[(kb * 16 + nt1) * 32 + lane];
            acc0 = __builtin_amdgcn_wmma_f32_16x16x32_f16(
                false, a, false, b0, (short)0, acc0, false, false);
            acc1 = __builtin_amdgcn_wmma_f32_16x16x32_f16(
                false, a, false, b1, (short)0, acc1, false, false);
        }

        // ---------- Phase 3: alpha_new = mask ? (max + log(acc) + emit) : alpha
        {
            const int mhalf = (lane >> 4) << 3;   // +8 rows for upper half-wave
            const int nl    = lane & 15;
            #pragma unroll
            for (int t = 0; t < 2; ++t) {
                const int n = (nt0 + t) * 16 + nl;
                #pragma unroll
                for (int v = 0; v < 8; ++v) {
                    int   row  = v + mhalf;
                    int   b    = m0 + row;
                    float accv = (t == 0) ? acc0[v] : acc1[v];
                    float emit = logits[(((size_t)b * S_DIM) + s) * T_DIM + n];
                    float upd  = mrowS[row] + __logf(accv) + emit;
                    float old  = alphaS[row * ASTRIDE + n];
                    int   mv   = mask[b * S_DIM + s];
                    alphaS[row * ASTRIDE + n] = mv ? upd : old;
                }
            }
        }
        __syncthreads();
    }

    // ---------- Final: den[b] = logsumexp(alpha[b,:]); out += num - den
    {
        int r  = tid >> 4;
        int cb = tid & 15;
        float mx = -__builtin_inff();
        for (int c = cb; c < T_DIM; c += 16)
            mx = fmaxf(mx, alphaS[r * ASTRIDE + c]);
        #pragma unroll
        for (int off = 1; off < 16; off <<= 1)
            mx = fmaxf(mx, __shfl_xor(mx, off, 32));
        float sum = 0.0f;
        for (int c = cb; c < T_DIM; c += 16)
            sum += __expf(alphaS[r * ASTRIDE + c] - mx);
        #pragma unroll
        for (int off = 1; off < 16; off <<= 1)
            sum += __shfl_xor(sum, off, 32);
        if (cb == 0) {
            float den = mx + __logf(sum);
            atomicAdd(out, ws_num[m0 + r] - den);
        }
    }
}

// =====================================================================
extern "C" void kernel_launch(void* const* d_in, const int* in_sizes, int n_in,
                              void* d_out, int out_size, void* d_ws, size_t ws_size,
                              hipStream_t stream) {
    const float* logits = (const float*)d_in[0];
    const float* trans  = (const float*)d_in[1];
    const int*   tags   = (const int*)d_in[2];
    const int*   mask   = (const int*)d_in[3];
    float* out    = (float*)d_out;
    float* ws_num = (float*)d_ws;   // 128 floats

    // Allow >64KB dynamic LDS (CDNA5 WGP has 320KB)
    (void)hipFuncSetAttribute(reinterpret_cast<const void*>(crf_fwd_kernel),
                              hipFuncAttributeMaxDynamicSharedMemorySize,
                              SMEM_BYTES);

    crf_num_init_kernel<<<1, 128, 0, stream>>>(logits, trans, tags, mask, out, ws_num);
    crf_fwd_kernel<<<B_DIM / 16, 256, SMEM_BYTES, stream>>>(logits, trans, mask, out, ws_num);
}
